// PointConv_12549894439208
// MI455X (gfx1250) — matrix operand order repaired and checked by reading, not compiled, not run on hardware
//
#include <hip/hip_runtime.h>

typedef __attribute__((ext_vector_type(16))) _Float16 v16h;
typedef __attribute__((ext_vector_type(8)))  float    v8f;

#define BLK_E 128
#define BLK_D 128

__device__ inline v8f wmma_f16(v16h a, v16h b, v8f c) {
  return __builtin_amdgcn_wmma_f32_16x16x32_f16(false, a, false, b, (short)0, c,
                                                false, false);
}

// A fragment (16x32 f16): row m = lane%16 ; K split per half-wave per ISA table:
// lanes 0-15: V0..3 -> K=0..7, V4..7 -> K=16..23 ; lanes 16-31: +8 within each half.
__device__ inline v16h load_a(const _Float16* tile, int ldk, int kbase, int lane) {
  const int m  = lane & 15;
  const int hh = (lane >> 4) & 1;
  union { v16h v; unsigned int u[8]; } f;
  const _Float16* p = tile + m * ldk + kbase;
#pragma unroll
  for (int r = 0; r < 8; ++r) {
    const int k0 = (r < 4) ? (hh * 8 + 2 * r) : (16 + hh * 8 + 2 * (r - 4));
    f.u[r] = *(const unsigned int*)(p + k0);
  }
  return f.v;
}

// B fragment (32x16 f16) from n-major (transposed) weights wt[N][K]:
// n = lane%16 ; VGPR v holds K = half*8 + {2v, 2v+1} for v<4, 16 + half*8 + {2(v-4), 2(v-4)+1} for v>=4 (dword loads).
__device__ inline v16h load_b(const _Float16* wt, int ldk, int n, int kbase, int lane) {
  const int hh = (lane >> 4) & 1;
  union { v16h v; unsigned int u[8]; } f;
  const _Float16* p = wt + n * ldk + kbase + hh * 8;
#pragma unroll
  for (int r = 0; r < 8; ++r) f.u[r] = *(const unsigned int*)(p + ((r < 4) ? 2 * r : 8 + 2 * r));
  return f.v;
}

// float atomic max via monotonic int/uint ordering; agg initialized to -inf.
__device__ inline void atomicMaxF32(float* addr, float val) {
  if (val >= 0.0f) {
    atomicMax((int*)addr, __float_as_int(val));
  } else {
    atomicMin((unsigned int*)addr, __float_as_uint(val));
  }
}

__global__ __launch_bounds__(256) void pc_init_kernel(unsigned int* agg, int n) {
  int i = blockIdx.x * blockDim.x + threadIdx.x;
  if (i < n) agg[i] = 0xFF800000u;  // -inf
}

// ---------------- Kernel 1: fused edge MLP + scatter-max -------------------
__global__ __launch_bounds__(256) void pc_edge_kernel(
    const float* __restrict__ pos, const float* __restrict__ pos_dst,
    const int* __restrict__ row, const int* __restrict__ col,
    const float* __restrict__ basis,
    const float* __restrict__ W1, const float* __restrict__ b1,
    const float* __restrict__ W2, const float* __restrict__ b2,
    float* __restrict__ agg, int nE) {
  __shared__ __align__(16) _Float16 feat[BLK_E][64];  // features, reused for h1
  __shared__ __align__(16) _Float16 wt1[64][64];      // W1^T, K padded 35->64
  __shared__ __align__(16) _Float16 wt2[64][64];      // W2^T
  __shared__ float sb1[64];
  __shared__ float sb2[64];
  __shared__ float sbas[48];
  __shared__ int   colT[BLK_E];

  const int tid  = threadIdx.x;
  const int lane = tid & 31;
  const int wave = tid >> 5;            // 8 waves, 16 edges each
  const int eb   = blockIdx.x * BLK_E;

  // Stage transposed f16 weights + biases + basis
  for (int i = tid; i < 64 * 64; i += 256) {
    int n = i >> 6, k = i & 63;
    wt1[n][k] = (_Float16)((k < 35) ? W1[k * 64 + n] : 0.0f);
    wt2[n][k] = (_Float16)W2[k * 64 + n];
  }
  if (tid < 64) { sb1[tid] = b1[tid]; sb2[tid] = b2[tid]; }
  if (tid < 48) sbas[tid] = basis[tid];
  __syncthreads();

  // Stage 128 edge features (one edge per thread 0..127)
  if (tid < BLK_E) {
    int e = eb + tid;
    if (e < nE) {
      int rs = row[e], cd = col[e];
      colT[tid] = cd;
      float rx = pos[rs * 3 + 0] - pos_dst[cd * 3 + 0];
      float ry = pos[rs * 3 + 1] - pos_dst[cd * 3 + 1];
      float rz = pos[rs * 3 + 2] - pos_dst[cd * 3 + 2];
      feat[tid][0] = (_Float16)rx;
      feat[tid][1] = (_Float16)ry;
      feat[tid][2] = (_Float16)rz;
#pragma unroll
      for (int j = 0; j < 16; ++j) {
        float emb = rx * sbas[j] + ry * sbas[16 + j] + rz * sbas[32 + j];
        feat[tid][3 + j]  = (_Float16)__sinf(emb);
        feat[tid][19 + j] = (_Float16)__cosf(emb);
      }
#pragma unroll
      for (int j = 35; j < 64; ++j) feat[tid][j] = (_Float16)0.0f;
    } else {
      colT[tid] = -1;
#pragma unroll
      for (int j = 0; j < 64; ++j) feat[tid][j] = (_Float16)0.0f;
    }
  }
  __syncthreads();

  const int nlane = lane & 15;
  const int mrow0 = (lane >> 4) * 8;           // D-matrix: VGPR r -> m = mrow0 + r
  const _Float16* atile = &feat[wave * 16][0];

  // ---- layer 1: [16x64] @ [64x64] ----
  v16h a0 = load_a(atile, 64, 0, lane);
  v16h a1 = load_a(atile, 64, 32, lane);
#pragma unroll
  for (int nt = 0; nt < 4; ++nt) {
    float bv = sb1[nt * 16 + nlane];
    v8f acc;
#pragma unroll
    for (int r = 0; r < 8; ++r) acc[r] = bv;
    acc = wmma_f16(a0, load_b(&wt1[0][0], 64, nt * 16 + nlane, 0,  lane), acc);
    acc = wmma_f16(a1, load_b(&wt1[0][0], 64, nt * 16 + nlane, 32, lane), acc);
#pragma unroll
    for (int r = 0; r < 8; ++r) {
      float v = acc[r] > 0.0f ? acc[r] : 0.0f;     // relu
      feat[wave * 16 + mrow0 + r][nt * 16 + nlane] = (_Float16)v;
    }
  }
  __syncthreads();

  // ---- layer 2: [16x64] @ [64x64], then scatter-max ----
  v16h h0 = load_a(atile, 64, 0, lane);
  v16h h1 = load_a(atile, 64, 32, lane);
#pragma unroll
  for (int nt = 0; nt < 4; ++nt) {
    float bv = sb2[nt * 16 + nlane];
    v8f acc;
#pragma unroll
    for (int r = 0; r < 8; ++r) acc[r] = bv;
    acc = wmma_f16(h0, load_b(&wt2[0][0], 64, nt * 16 + nlane, 0,  lane), acc);
    acc = wmma_f16(h1, load_b(&wt2[0][0], 64, nt * 16 + nlane, 32, lane), acc);
#pragma unroll
    for (int r = 0; r < 8; ++r) {
      int c = colT[wave * 16 + mrow0 + r];
      if (c >= 0) atomicMaxF32(&agg[(long)c * 64 + nt * 16 + nlane], acc[r]);
    }
  }
}

// ---------------- Kernel 2: destination MLP 64->128->128 -------------------
__global__ __launch_bounds__(256) void pc_dst_kernel(
    const float* __restrict__ agg,
    const float* __restrict__ W3, const float* __restrict__ b3,
    const float* __restrict__ W4, const float* __restrict__ b4,
    float* __restrict__ out, int nM) {
  __shared__ __align__(16) _Float16 wt3[128][64];    // W3^T  (16 KB)
  __shared__ __align__(16) _Float16 wt4[128][128];   // W4^T  (32 KB)
  __shared__ __align__(16) _Float16 aT[BLK_D][64];   // agg tile (16 KB)
  __shared__ __align__(16) _Float16 hT[BLK_D][128];  // hidden (32 KB)
  __shared__ float sb3[128];
  __shared__ float sb4[128];

  const int tid  = threadIdx.x;
  const int lane = tid & 31;
  const int wave = tid >> 5;
  const int db   = blockIdx.x * BLK_D;

  for (int i = tid; i < 128 * 64; i += 256) {
    int n = i >> 6, k = i & 63;
    wt3[n][k] = (_Float16)W3[k * 128 + n];
  }
  for (int i = tid; i < 128 * 128; i += 256) {
    int n = i >> 7, k = i & 127;
    wt4[n][k] = (_Float16)W4[k * 128 + n];
  }
  if (tid < 128) { sb3[tid] = b3[tid]; sb4[tid] = b4[tid]; }

  // Stage agg rows for this wave (16 rows x 64 cols), -inf -> 0
  {
    const int r0 = lane >> 1;
    const int c0 = (lane & 1) * 32;
    int dst = db + wave * 16 + r0;
#pragma unroll
    for (int j = 0; j < 32; ++j) {
      float v = 0.0f;
      if (dst < nM) {
        v = agg[(long)dst * 64 + c0 + j];
        if (v == -__builtin_inff()) v = 0.0f;   // empty segments -> 0
      }
      aT[wave * 16 + r0][c0 + j] = (_Float16)v;
    }
  }
  __syncthreads();

  const int nlane = lane & 15;
  const int mrow0 = (lane >> 4) * 8;

  // ---- layer 3: [16x64] @ [64x128] + relu ----
  {
    const _Float16* atile = &aT[wave * 16][0];
    v16h a0 = load_a(atile, 64, 0, lane);
    v16h a1 = load_a(atile, 64, 32, lane);
#pragma unroll
    for (int nt = 0; nt < 8; ++nt) {
      float bv = sb3[nt * 16 + nlane];
      v8f acc;
#pragma unroll
      for (int r = 0; r < 8; ++r) acc[r] = bv;
      acc = wmma_f16(a0, load_b(&wt3[0][0], 64, nt * 16 + nlane, 0,  lane), acc);
      acc = wmma_f16(a1, load_b(&wt3[0][0], 64, nt * 16 + nlane, 32, lane), acc);
#pragma unroll
      for (int r = 0; r < 8; ++r) {
        float v = acc[r] > 0.0f ? acc[r] : 0.0f;
        hT[wave * 16 + mrow0 + r][nt * 16 + nlane] = (_Float16)v;
      }
    }
  }
  __syncthreads();

  // ---- layer 4: [16x128] @ [128x128] -> out ----
  {
    const _Float16* htile = &hT[wave * 16][0];
    v16h ha[4];
#pragma unroll
    for (int ks = 0; ks < 4; ++ks) ha[ks] = load_a(htile, 128, ks * 32, lane);
#pragma unroll
    for (int nt = 0; nt < 8; ++nt) {
      float bv = sb4[nt * 16 + nlane];
      v8f acc;
#pragma unroll
      for (int r = 0; r < 8; ++r) acc[r] = bv;
#pragma unroll
      for (int ks = 0; ks < 4; ++ks)
        acc = wmma_f16(ha[ks], load_b(&wt4[0][0], 128, nt * 16 + nlane, ks * 32, lane), acc);
#pragma unroll
      for (int r = 0; r < 8; ++r) {
        int dst = db + wave * 16 + mrow0 + r;
        if (dst < nM) out[(long)dst * 128 + nt * 16 + nlane] = acc[r];
      }
    }
  }
}

extern "C" void kernel_launch(void* const* d_in, const int* in_sizes, int n_in,
                              void* d_out, int out_size, void* d_ws, size_t ws_size,
                              hipStream_t stream) {
  const float* pos     = (const float*)d_in[0];
  const float* pos_dst = (const float*)d_in[1];
  const int*   row     = (const int*)d_in[2];
  const int*   col     = (const int*)d_in[3];
  const float* basis   = (const float*)d_in[4];
  const float* W1      = (const float*)d_in[5];
  const float* b1      = (const float*)d_in[6];
  const float* W2      = (const float*)d_in[7];
  const float* b2      = (const float*)d_in[8];
  const float* W3      = (const float*)d_in[9];
  const float* b3      = (const float*)d_in[10];
  const float* W4      = (const float*)d_in[11];
  const float* b4      = (const float*)d_in[12];
  float* out = (float*)d_out;
  float* agg = (float*)d_ws;                 // [M, 64] f32 scratch (25.6 MB)

  const int nM = in_sizes[1] / 3;            // 100000
  const int nE = in_sizes[2];                // 1600000

  const int aggN = nM * 64;
  pc_init_kernel<<<(aggN + 255) / 256, 256, 0, stream>>>((unsigned int*)agg, aggN);
  pc_edge_kernel<<<(nE + BLK_E - 1) / BLK_E, 256, 0, stream>>>(
      pos, pos_dst, row, col, basis, W1, b1, W2, b2, agg, nE);
  pc_dst_kernel<<<(nM + BLK_D - 1) / BLK_D, 256, 0, stream>>>(
      agg, W3, b3, W4, b4, out, nM);
}
